// CrossViewSwapAttention_498216206553
// MI455X (gfx1250) — matrix operand order repaired and checked by reading, not compiled, not run on hardware
//
#include <hip/hip_runtime.h>
#include <math.h>

// ---------------------------------------------------------------------------
// CrossViewSwapAttention forward for MI455X (gfx1250, wave32, WMMA).
// All GEMM-shaped work uses v_wmma_f32_16x16x32_f16 (f16 in, f32 accum).
// Weights are pre-converted to f16 (B-transposed) so all WMMA operand loads
// are contiguous 16-byte vector loads; K/V are written window-gathered so the
// attention kernel streams them with b128 loads.
// ---------------------------------------------------------------------------

#define DIMC   128
#define NHEADS 4
#define DHD    32
#define HQ     64
#define WQ     64
#define FHH    28
#define FWW    60
#define NVIEW  6
#define NPIX   (FHH * FWW)        // 1680
#define PH     32
#define PW     64
#define PPIX   (PH * PW)          // 2048 (padded 32x64)
#define TQ1    (NVIEW * HQ * WQ)  // 24576 pass-1 query tokens
#define TKV    (NVIEW * PPIX)     // 12288 key/val tokens
#define TS     (HQ * WQ)          // 4096 spatial tokens
#define NWIN   16
#define KWIN   768                // key tokens per window (6 views * 128)
#define SCALE_ATT 0.17677669529663687f  // 32^-0.5

typedef __attribute__((ext_vector_type(16))) _Float16 v16h;
typedef __attribute__((ext_vector_type(8)))  _Float16 v8h;
typedef __attribute__((ext_vector_type(8)))  float    v8f;

// ---- WMMA fragment lane layouts (CDNA5 ISA 7.12.2, wave32) ----------------
// A (16x32 f16): row m = lane&15; elem e -> k = ((e<8)?e:e+8) + ((lane&16)?8:0)
// B (32x16 f16): col n = lane&15; elem e -> k = ((e<8)?e:e+8) + ((lane&16)?8:0)
// C/D (16x16 f32): vgpr r -> row m = r + ((lane&16)?8:0), col n = lane&15.
__device__ __forceinline__ int a_frag_k(int lane, int e) {
    return ((e < 8) ? e : e + 8) + ((lane & 16) ? 8 : 0);
}
__device__ __forceinline__ int b_frag_k(int lane, int e) {
    return ((e < 8) ? e : e + 8) + ((lane & 16) ? 8 : 0);
}

__device__ __forceinline__ v16h cat16(v8h lo, v8h hi) {
    v16h r;
    #pragma unroll
    for (int i = 0; i < 8; ++i) { r[i] = lo[i]; r[i + 8] = hi[i]; }
    return r;
}
// Contiguous A-fragment load: row points at the 32 f16 K-values for this row.
__device__ __forceinline__ v16h load_a(const _Float16* row, int lane) {
    int off = (lane & 16) ? 8 : 0;
    return cat16(*(const v8h*)(row + off), *(const v8h*)(row + off + 16));
}
// Contiguous B-fragment load: col points at the 32 f16 K-values for this col.
__device__ __forceinline__ v16h load_b(const _Float16* col, int lane) {
    int off = (lane & 16) ? 8 : 0;
    return cat16(*(const v8h*)(col + off), *(const v8h*)(col + off + 16));
}

__device__ __forceinline__ v8f wmma_f16(v16h a, v16h b, v8f c) {
    return __builtin_amdgcn_wmma_f32_16x16x32_f16(false, a, false, b,
                                                  (short)0, c, false, false);
}

// ---------------------------------------------------------------------------
// Weight prep: f32 (K x N) -> f16, optionally transposed to (N x K).
// ---------------------------------------------------------------------------
__global__ __launch_bounds__(256) void k_wprep(const float* __restrict__ W,
                                               int K, int N, int transpose,
                                               _Float16* __restrict__ out) {
    int idx = blockIdx.x * 256 + threadIdx.x;
    if (idx >= K * N) return;
    int k = idx / N, n = idx % N;
    _Float16 v = (_Float16)W[idx];
    if (transpose) out[(size_t)n * K + k] = v;
    else           out[idx] = v;
}

// ---------------------------------------------------------------------------
// Camera embedding  c_embed[v][c] = sum_j cam_w[c][j] * E_inv[v][j][3]
// ---------------------------------------------------------------------------
__global__ __launch_bounds__(768) void k_cembed(const float* __restrict__ E_inv,
                                                const float* __restrict__ cam_w,
                                                float* __restrict__ cembed) {
    int i = threadIdx.x;
    int v = i >> 7, c = i & 127;
    const float* Ev = E_inv + v * 16;
    float s = 0.f;
    #pragma unroll
    for (int j = 0; j < 4; ++j) s += cam_w[c * 4 + j] * Ev[j * 4 + 3];
    cembed[i] = s;
}

// ---------------------------------------------------------------------------
// Image-plane embedding (normalized over channels)
// ---------------------------------------------------------------------------
__global__ __launch_bounds__(128) void k_img_embed(const float* __restrict__ I_inv,
                                                   const float* __restrict__ E_inv,
                                                   const float* __restrict__ img_w,
                                                   const float* __restrict__ cembed,
                                                   float* __restrict__ img_e) {
    int blk = blockIdx.x;
    int v = blk / NPIX, pix = blk % NPIX;
    int fy = pix / FWW, fx = pix % FWW;
    float px = fx * (480.0f / 59.0f);
    float py = fy * (224.0f / 27.0f);
    const float* Iv = I_inv + v * 9;
    const float* Ev = E_inv + v * 16;
    float c0 = Iv[0] * px + Iv[1] * py + Iv[2];
    float c1 = Iv[3] * px + Iv[4] * py + Iv[5];
    float c2 = Iv[6] * px + Iv[7] * py + Iv[8];
    float d[4];
    #pragma unroll
    for (int i = 0; i < 4; ++i)
        d[i] = Ev[i * 4 + 0] * c0 + Ev[i * 4 + 1] * c1 + Ev[i * 4 + 2] * c2 + Ev[i * 4 + 3];
    int c = threadIdx.x;
    float e = img_w[c * 4 + 0] * d[0] + img_w[c * 4 + 1] * d[1] +
              img_w[c * 4 + 2] * d[2] + img_w[c * 4 + 3] * d[3] - cembed[v * 128 + c];
    __shared__ float red[128];
    red[c] = e * e; __syncthreads();
    for (int s = 64; s > 0; s >>= 1) { if (c < s) red[c] += red[c + s]; __syncthreads(); }
    float nrm = sqrtf(red[0]) + 1e-7f;
    img_e[((size_t)(v * DIMC + c)) * NPIX + pix] = e / nrm;
}

// ---------------------------------------------------------------------------
// BEV embedding -> query_pos (6,128,64,64), normalized
// ---------------------------------------------------------------------------
__global__ __launch_bounds__(128) void k_bev_embed(const float* __restrict__ bev,
                                                   const float* __restrict__ bev_w,
                                                   const float* __restrict__ bev_b,
                                                   const float* __restrict__ cembed,
                                                   float* __restrict__ qpos) {
    int blk = blockIdx.x;
    int v = blk / TS, p = blk % TS;
    int c = threadIdx.x;
    float e = bev_w[c * 2 + 0] * bev[p] + bev_w[c * 2 + 1] * bev[TS + p] +
              bev_b[c] - cembed[v * 128 + c];
    __shared__ float red[128];
    red[c] = e * e; __syncthreads();
    for (int s = 64; s > 0; s >>= 1) { if (c < s) red[c] += red[c + s]; __syncthreads(); }
    float nrm = sqrtf(red[0]) + 1e-7f;
    qpos[((size_t)(v * DIMC + c)) * TS + p] = e / nrm;
}

// ---------------------------------------------------------------------------
// BatchNorm stats + fold into per-channel scale/shift for both projections.
// ---------------------------------------------------------------------------
__global__ __launch_bounds__(256) void k_bn_stats(const float* __restrict__ feat,
                                                  const float* __restrict__ gfp,
                                                  const float* __restrict__ bfp,
                                                  const float* __restrict__ gfl,
                                                  const float* __restrict__ bfl,
                                                  float* __restrict__ sfp,
                                                  float* __restrict__ ofp,
                                                  float* __restrict__ sfl,
                                                  float* __restrict__ ofl) {
    int c = blockIdx.x, t = threadIdx.x;
    float s1 = 0.f, s2 = 0.f;
    for (int v = 0; v < NVIEW; ++v) {
        const float* p = feat + ((size_t)(v * DIMC + c)) * NPIX;
        for (int i = t; i < NPIX; i += 256) { float f = p[i]; s1 += f; s2 += f * f; }
    }
    __shared__ float r1[256], r2[256];
    r1[t] = s1; r2[t] = s2; __syncthreads();
    for (int s = 128; s > 0; s >>= 1) {
        if (t < s) { r1[t] += r1[t + s]; r2[t] += r2[t + s]; }
        __syncthreads();
    }
    if (t == 0) {
        const float inv = 1.0f / (NVIEW * NPIX);
        float m = r1[0] * inv;
        float var = r2[0] * inv - m * m;
        float rs = rsqrtf(var + 1e-5f);
        float a = gfp[c] * rs; sfp[c] = a; ofp[c] = bfp[c] - m * a;
        float b = gfl[c] * rs; sfl[c] = b; ofl[c] = bfl[c] - m * b;
    }
}

// ---------------------------------------------------------------------------
// BN+ReLU+1x1conv for key/val, padded to 32x64; key gets img_e added.
// One wave per 16x16 output tile; branch-free padded loads. WMMA.
// ---------------------------------------------------------------------------
__global__ __launch_bounds__(32) void k_featproj(const float* __restrict__ feat,
                                                 const float* __restrict__ sfp,
                                                 const float* __restrict__ ofp,
                                                 const _Float16* __restrict__ wfp,
                                                 const float* __restrict__ sfl,
                                                 const float* __restrict__ ofl,
                                                 const _Float16* __restrict__ wfl,
                                                 const float* __restrict__ img_e,
                                                 float* __restrict__ key_pad,
                                                 float* __restrict__ val_pad) {
    int lane = threadIdx.x;
    int nt = blockIdx.x, mt = blockIdx.y, v = blockIdx.z;
    int n0 = nt * 16, m0 = mt * 16;
    int lm = lane & 15, hi = (lane & 16) ? 1 : 0;
    int pp = n0 + lm;                 // padded pixel for this lane's B column
    int hh = pp >> 6, ww = pp & 63;
    bool valid = (hh < FHH) && (ww < FWW);
    int pix = valid ? (hh * FWW + ww) : 0;   // clamped: unconditional loads
    float fmask = valid ? 1.0f : 0.0f;

    v8f ack = {}, acv = {};
    for (int kc = 0; kc < DIMC; kc += 32) {
        v16h aK = load_a(wfp + (m0 + lm) * DIMC + kc, lane);
        v16h aV = load_a(wfl + (m0 + lm) * DIMC + kc, lane);
        v16h bK, bV;
        #pragma unroll
        for (int e = 0; e < 16; ++e) {
            int kb = kc + b_frag_k(lane, e);
            float f = feat[((size_t)(v * DIMC + kb)) * NPIX + pix];
            bK[e] = (_Float16)(fmaxf(f * sfp[kb] + ofp[kb], 0.f) * fmask);
            bV[e] = (_Float16)(fmaxf(f * sfl[kb] + ofl[kb], 0.f) * fmask);
        }
        ack = wmma_f16(aK, bK, ack);
        acv = wmma_f16(aV, bV, acv);
    }
    #pragma unroll
    for (int r = 0; r < 8; ++r) {
        int mm = m0 + r + hi * 8;
        size_t o = ((size_t)(v * DIMC + mm)) * PPIX + pp;
        float imge = img_e[((size_t)(v * DIMC + mm)) * NPIX + pix];
        key_pad[o] = valid ? (imge + ack[r]) : 0.f;
        val_pad[o] = valid ? acv[r] : 0.f;
    }
}

// ---------------------------------------------------------------------------
// LayerNorm over 128 channels -> f16 token-major output.
// mode 0: t=(v,p) val = qpos[(v*128+c)*4096+p] + x[c*4096+p]
// mode 1: t=(v,p) val = src0[(v*128+c)*2048+p]        (padded key/val)
// mode 2: token-major src0[t*128+c]
// ---------------------------------------------------------------------------
__global__ __launch_bounds__(128) void k_ln_tok(const float* __restrict__ src0,
                                                const float* __restrict__ src1,
                                                const float* __restrict__ g,
                                                const float* __restrict__ b,
                                                int mode,
                                                _Float16* __restrict__ out) {
    int t = blockIdx.x, c = threadIdx.x;
    float val;
    if (mode == 0) {
        int v = t >> 12, p = t & 4095;
        val = src0[((size_t)(v * DIMC + c)) * TS + p] + src1[(size_t)c * TS + p];
    } else if (mode == 1) {
        int v = t >> 11, p = t & 2047;
        val = src0[((size_t)(v * DIMC + c)) * PPIX + p];
    } else {
        val = src0[(size_t)t * DIMC + c];
    }
    __shared__ float red[128];
    red[c] = val; __syncthreads();
    for (int s = 64; s > 0; s >>= 1) { if (c < s) red[c] += red[c + s]; __syncthreads(); }
    float m = red[0] * (1.0f / 128.0f); __syncthreads();
    float dv = val - m;
    red[c] = dv * dv; __syncthreads();
    for (int s = 64; s > 0; s >>= 1) { if (c < s) red[c] += red[c + s]; __syncthreads(); }
    float inv = rsqrtf(red[0] * (1.0f / 128.0f) + 1e-5f);
    out[(size_t)t * DIMC + c] = (_Float16)(dv * inv * g[c] + b[c]);
}

// ---------------------------------------------------------------------------
// Destination permutation for K/V projection outputs.
// kv_mode 0: out[t*N+col]
// kv_mode 1/2: K pass1/pass2 window-gathered rows  [win*768+kl][128]
// kv_mode 3/4: V pass1/pass2 transposed per window [win*128+col][768]
// ---------------------------------------------------------------------------
__device__ __forceinline__ size_t kv_dest(int kv_mode, int t, int col, int N) {
    if (kv_mode == 0) return (size_t)t * N + col;
    int v = t >> 11, pp = t & 2047;
    int hh = pp >> 6, ww = pp & 63;
    int X, Y, i, j;
    if (kv_mode == 1 || kv_mode == 3) { X = hh >> 3; i = hh & 7;  Y = ww >> 4; j = ww & 15; }
    else                              { X = hh & 3;  i = hh >> 2; Y = ww & 3;  j = ww >> 2; }
    int win = X * 4 + Y;
    int kl = v * 128 + i * 16 + j;
    if (kv_mode <= 2) return ((size_t)win * KWIN + kl) * DIMC + col;
    return ((size_t)win * DIMC + col) * KWIN + kl;
}

// ---------------------------------------------------------------------------
// Generic token GEMM  out = act(scale*(Xn@W) + bias) + skip
// Xn: T x K f16 token-major; Wt: N x K f16 (pre-transposed).
// One wave per 16x16 tile; all WMMA operand loads are b128 vector loads.
// ---------------------------------------------------------------------------
__global__ __launch_bounds__(32) void k_gemm_tok(const _Float16* __restrict__ Xn,
                                                 const _Float16* __restrict__ Wt,
                                                 const float* __restrict__ bias,
                                                 float scale,
                                                 const float* __restrict__ skip,
                                                 int skip_cm,   // 0: [t*N+n], 1: [n*T+t]
                                                 int T, int K, int N,
                                                 int act,       // 0 none, 1 exact GELU
                                                 int kv_mode,
                                                 float* __restrict__ out_f32,
                                                 _Float16* __restrict__ out_f16) {
    int lane = threadIdx.x;
    int mt = blockIdx.x, nt = blockIdx.y;
    int t0 = mt * 16, n0 = nt * 16;
    int lm = lane & 15, hi = (lane & 16) ? 1 : 0;
    const _Float16* arow = Xn + (size_t)(t0 + lm) * K;
    const _Float16* bcol = Wt + (size_t)(n0 + lm) * K;
    v8f acc = {};
    for (int kc = 0; kc < K; kc += 32) {
        v16h a = load_a(arow + kc, lane);
        v16h b = load_b(bcol + kc, lane);
        acc = wmma_f16(a, b, acc);
    }
    #pragma unroll
    for (int r = 0; r < 8; ++r) {
        int mm = t0 + r + hi * 8;
        int col = n0 + lm;
        float vv = acc[r] * scale;
        if (bias) vv += bias[col];
        if (act == 1) vv = 0.5f * vv * (1.0f + erff(vv * 0.70710678118654752f));
        if (skip) vv += skip_cm ? skip[(size_t)col * T + mm] : skip[(size_t)mm * N + col];
        size_t o = kv_dest(kv_mode, mm, col, N);
        if (out_f32) out_f32[o] = vv;
        if (out_f16) out_f16[o] = (_Float16)vv;
    }
}

// ---------------------------------------------------------------------------
// Windowed cross attention. One wave per (16-query tile, head, window).
// Kw: window-gathered rows [win*768+kl][128]; Vt: per-window transposed
// [win*128+c][768]. S(16x768) staged in LDS; softmax; P@V with WMMA.
// ---------------------------------------------------------------------------
__global__ __launch_bounds__(32) void k_attn(const _Float16* __restrict__ Qp,
                                             const _Float16* __restrict__ Kw,
                                             const _Float16* __restrict__ Vt,
                                             float* __restrict__ att,
                                             int pass) {
    __shared__ float S[16 * KWIN];
    __shared__ float rmax[16], rsum[16];
    int lane = threadIdx.x;
    int qt = blockIdx.x, head = blockIdx.y, win = blockIdx.z;
    int X = win >> 2, Y = win & 3;
    int ho = head * DHD;
    int lm = lane & 15, hi = (lane & 16) ? 1 : 0;

    // Query token index for A-fragment row lm
    int tq_m;
    {
        int ql = qt * 16 + lm;
        if (pass == 1) {
            int v = ql >> 8, r = ql & 255;
            tq_m = v * TS + (X * 16 + (r >> 4)) * WQ + Y * 16 + (r & 15);
        } else {
            tq_m = (X * 16 + (ql >> 4)) * WQ + Y * 16 + (ql & 15);
        }
    }
    v16h aq = load_a(Qp + (size_t)tq_m * DIMC + ho, lane);

    // Phase A: S = scale * Q @ K^T  (48 WMMAs -> LDS), contiguous K loads
    const _Float16* kbase = Kw + ((size_t)win * KWIN + lm) * DIMC + ho;
    for (int kt = 0; kt < 48; ++kt) {
        v16h bk = load_b(kbase + (size_t)kt * 16 * DIMC, lane);
        v8f c = {};
        c = wmma_f16(aq, bk, c);
        #pragma unroll
        for (int r = 0; r < 8; ++r)
            S[(r + hi * 8) * KWIN + kt * 16 + lm] = c[r] * SCALE_ATT;
    }

    // Phase B: row softmax stats, all 32 lanes + cross-half shuffle combine
    {
        const float* Sr = &S[lm * KWIN + hi * 384];
        float mx = -1e30f;
        for (int k = 0; k < 384; ++k) mx = fmaxf(mx, Sr[k]);
        mx = fmaxf(mx, __shfl_xor(mx, 16, 32));
        float sm = 0.f;
        for (int k = 0; k < 384; ++k) sm += expf(Sr[k] - mx);
        sm += __shfl_xor(sm, 16, 32);
        if (!hi) { rmax[lm] = mx; rsum[lm] = sm; }
    }

    // Phase C: O = P @ V (unnormalized), contiguous Vt loads
    const _Float16* v0 = Vt + ((size_t)win * DIMC + ho + lm) * KWIN;
    const _Float16* v1 = Vt + ((size_t)win * DIMC + ho + 16 + lm) * KWIN;
    v8f o0 = {}, o1 = {};
    for (int kc = 0; kc < 24; ++kc) {
        v16h ap;
        #pragma unroll
        for (int e = 0; e < 16; ++e) {
            int kk = kc * 32 + a_frag_k(lane, e);
            ap[e] = (_Float16)expf(S[lm * KWIN + kk] - rmax[lm]);
        }
        v16h bv0 = load_b(v0 + kc * 32, lane);
        v16h bv1 = load_b(v1 + kc * 32, lane);
        o0 = wmma_f16(ap, bv0, o0);
        o1 = wmma_f16(ap, bv1, o1);
    }
    #pragma unroll
    for (int r = 0; r < 8; ++r) {
        int mm = r + hi * 8;
        int ql = qt * 16 + mm;
        int tq;
        if (pass == 1) {
            int v = ql >> 8, rr = ql & 255;
            tq = v * TS + (X * 16 + (rr >> 4)) * WQ + Y * 16 + (rr & 15);
        } else {
            tq = (X * 16 + (ql >> 4)) * WQ + Y * 16 + (ql & 15);
        }
        float inv = 1.0f / rsum[mm];
        att[(size_t)tq * DIMC + ho + lm]      = o0[r] * inv;
        att[(size_t)tq * DIMC + ho + 16 + lm] = o1[r] * inv;
    }
}

// ---------------------------------------------------------------------------
// Sum attention output over views -> f16 tokens (nv=1 is a convert)
// ---------------------------------------------------------------------------
__global__ __launch_bounds__(128) void k_viewsum(const float* __restrict__ att,
                                                 int nv, _Float16* __restrict__ out) {
    int s = blockIdx.x, c = threadIdx.x;
    float a = 0.f;
    for (int v = 0; v < nv; ++v) a += att[((size_t)v * TS + s) * DIMC + c];
    out[(size_t)s * DIMC + c] = (_Float16)a;
}

// ---------------------------------------------------------------------------
// Final LayerNorm + transpose to (1,128,64,64)
// ---------------------------------------------------------------------------
__global__ __launch_bounds__(128) void k_post_ln(const float* __restrict__ qm2,
                                                 const float* __restrict__ g,
                                                 const float* __restrict__ b,
                                                 float* __restrict__ out) {
    int t = blockIdx.x, c = threadIdx.x;
    float val = qm2[(size_t)t * DIMC + c];
    __shared__ float red[128];
    red[c] = val; __syncthreads();
    for (int s = 64; s > 0; s >>= 1) { if (c < s) red[c] += red[c + s]; __syncthreads(); }
    float m = red[0] * (1.0f / 128.0f); __syncthreads();
    float dv = val - m;
    red[c] = dv * dv; __syncthreads();
    for (int s = 64; s > 0; s >>= 1) { if (c < s) red[c] += red[c + s]; __syncthreads(); }
    float inv = rsqrtf(red[0] * (1.0f / 128.0f) + 1e-5f);
    out[(size_t)c * TS + t] = dv * inv * g[c] + b[c];
}

// ---------------------------------------------------------------------------
// Input index map (setup_inputs() dict insertion order, params flattened
// depth-first in construction order).
// ---------------------------------------------------------------------------
enum {
    IN_X = 0, IN_FEAT, IN_IINV, IN_EINV, IN_BEV,
    P_CAM_W, P_IMG_W, P_BEV_W, P_BEV_B,
    P_FP_G, P_FP_B, P_FP_W, P_FL_G, P_FL_B, P_FL_W,
    A1_QN_G, A1_QN_B, A1_Q_W, A1_Q_B, A1_KN_G, A1_KN_B, A1_K_W, A1_K_B,
    A1_VN_G, A1_VN_B, A1_V_W, A1_V_B, A1_P_W, A1_P_B,
    A2_QN_G, A2_QN_B, A2_Q_W, A2_Q_B, A2_KN_G, A2_KN_B, A2_K_W, A2_K_B,
    A2_VN_G, A2_VN_B, A2_V_W, A2_V_B, A2_P_W, A2_P_B,
    P_PN1_G, P_PN1_B, P_M1W1, P_M1B1, P_M1W2, P_M1B2,
    P_PN2_G, P_PN2_B, P_M2W1, P_M2B1, P_M2W2, P_M2B2,
    P_POST_G, P_POST_B
};

extern "C" void kernel_launch(void* const* d_in, const int* in_sizes, int n_in,
                              void* d_out, int out_size, void* d_ws, size_t ws_size,
                              hipStream_t stream) {
    (void)in_sizes; (void)n_in; (void)out_size; (void)ws_size;
    auto F = [&](int i) { return (const float*)d_in[i]; };

    // ---- workspace layout (256B aligned) ----
    char* base = (char*)d_ws;
    size_t off = 0;
    auto alloc = [&](size_t bytes) -> char* {
        char* p = base + off;
        off += (bytes + 255) & ~(size_t)255;
        return p;
    };
    float*     ws_cembed = (float*)alloc(NVIEW * DIMC * 4);
    float*     ws_sfp    = (float*)alloc(DIMC * 4);
    float*     ws_ofp    = (float*)alloc(DIMC * 4);
    float*     ws_sfl    = (float*)alloc(DIMC * 4);
    float*     ws_ofl    = (float*)alloc(DIMC * 4);
    float*     ws_imge   = (float*)alloc((size_t)NVIEW * DIMC * NPIX * 4);
    float*     ws_qpos   = (float*)alloc((size_t)NVIEW * DIMC * TS * 4);
    float*     ws_keyp   = (float*)alloc((size_t)NVIEW * DIMC * PPIX * 4);
    float*     ws_valp   = (float*)alloc((size_t)NVIEW * DIMC * PPIX * 4);
    _Float16*  ws_xnq    = (_Float16*)alloc((size_t)TQ1 * DIMC * 2);
    _Float16*  ws_xnk    = (_Float16*)alloc((size_t)TKV * DIMC * 2);
    _Float16*  ws_xnv    = (_Float16*)alloc((size_t)TKV * DIMC * 2);
    _Float16*  ws_Qp     = (_Float16*)alloc((size_t)TQ1 * DIMC * 2);
    _Float16*  ws_Kw     = (_Float16*)alloc((size_t)TKV * DIMC * 2);
    _Float16*  ws_Vt     = (_Float16*)alloc((size_t)TKV * DIMC * 2);
    float*     ws_att    = (float*)alloc((size_t)TQ1 * DIMC * 4);
    _Float16*  ws_attsum = (_Float16*)alloc((size_t)TS * DIMC * 2);
    float*     ws_qm     = (float*)alloc((size_t)TS * DIMC * 4);
    float*     ws_qm2    = (float*)alloc((size_t)TS * DIMC * 4);
    _Float16*  ws_xnt    = (_Float16*)alloc((size_t)TS * DIMC * 2);
    _Float16*  ws_mlph   = (_Float16*)alloc((size_t)TS * 256 * 2);
    // f16 weights
    _Float16*  w_fp   = (_Float16*)alloc((size_t)DIMC * DIMC * 2);  // row-major
    _Float16*  w_fl   = (_Float16*)alloc((size_t)DIMC * DIMC * 2);  // row-major
    _Float16*  w_a1q  = (_Float16*)alloc((size_t)DIMC * DIMC * 2);  // transposed
    _Float16*  w_a1k  = (_Float16*)alloc((size_t)DIMC * DIMC * 2);
    _Float16*  w_a1v  = (_Float16*)alloc((size_t)DIMC * DIMC * 2);
    _Float16*  w_a1p  = (_Float16*)alloc((size_t)DIMC * DIMC * 2);
    _Float16*  w_a2q  = (_Float16*)alloc((size_t)DIMC * DIMC * 2);
    _Float16*  w_a2k  = (_Float16*)alloc((size_t)DIMC * DIMC * 2);
    _Float16*  w_a2v  = (_Float16*)alloc((size_t)DIMC * DIMC * 2);
    _Float16*  w_a2p  = (_Float16*)alloc((size_t)DIMC * DIMC * 2);
    _Float16*  w_m1a  = (_Float16*)alloc((size_t)DIMC * 256 * 2);
    _Float16*  w_m1b  = (_Float16*)alloc((size_t)256 * DIMC * 2);
    _Float16*  w_m2a  = (_Float16*)alloc((size_t)DIMC * 256 * 2);
    _Float16*  w_m2b  = (_Float16*)alloc((size_t)256 * DIMC * 2);

    auto wprep = [&](int pidx, int K, int N, int tr, _Float16* dst) {
        k_wprep<<<(K * N + 255) / 256, 256, 0, stream>>>(F(pidx), K, N, tr, dst);
    };
    auto gemm = [&](const _Float16* Xn, const _Float16* Wt, const float* bias,
                    float scale, const float* skip, int skip_cm,
                    int T, int K, int N, int act, int kv_mode,
                    float* of32, _Float16* of16) {
        dim3 g(T / 16, N / 16);
        k_gemm_tok<<<g, 32, 0, stream>>>(Xn, Wt, bias, scale, skip, skip_cm,
                                         T, K, N, act, kv_mode, of32, of16);
    };

    // ---- weight conversion ----
    wprep(P_FP_W, DIMC, DIMC, 0, w_fp);   // used as A (row-major [o][c])
    wprep(P_FL_W, DIMC, DIMC, 0, w_fl);
    wprep(A1_Q_W, DIMC, DIMC, 1, w_a1q);
    wprep(A1_K_W, DIMC, DIMC, 1, w_a1k);
    wprep(A1_V_W, DIMC, DIMC, 1, w_a1v);
    wprep(A1_P_W, DIMC, DIMC, 1, w_a1p);
    wprep(A2_Q_W, DIMC, DIMC, 1, w_a2q);
    wprep(A2_K_W, DIMC, DIMC, 1, w_a2k);
    wprep(A2_V_W, DIMC, DIMC, 1, w_a2v);
    wprep(A2_P_W, DIMC, DIMC, 1, w_a2p);
    wprep(P_M1W1, DIMC, 256, 1, w_m1a);
    wprep(P_M1W2, 256, DIMC, 1, w_m1b);
    wprep(P_M2W1, DIMC, 256, 1, w_m2a);
    wprep(P_M2W2, 256, DIMC, 1, w_m2b);

    // ---- stage 0: embeddings / BN ----
    k_cembed<<<1, 768, 0, stream>>>(F(IN_EINV), F(P_CAM_W), ws_cembed);
    k_img_embed<<<NVIEW * NPIX, 128, 0, stream>>>(F(IN_IINV), F(IN_EINV),
                                                  F(P_IMG_W), ws_cembed, ws_imge);
    k_bev_embed<<<NVIEW * TS, 128, 0, stream>>>(F(IN_BEV), F(P_BEV_W), F(P_BEV_B),
                                                ws_cembed, ws_qpos);
    k_bn_stats<<<DIMC, 256, 0, stream>>>(F(IN_FEAT), F(P_FP_G), F(P_FP_B),
                                         F(P_FL_G), F(P_FL_B),
                                         ws_sfp, ws_ofp, ws_sfl, ws_ofl);
    k_featproj<<<dim3(PPIX / 16, DIMC / 16, NVIEW), 32, 0, stream>>>(
        F(IN_FEAT), ws_sfp, ws_ofp, w_fp, ws_sfl, ws_ofl, w_fl,
        ws_imge, ws_keyp, ws_valp);

    // ---- pass 1 ----
    k_ln_tok<<<TQ1, 128, 0, stream>>>(ws_qpos, F(IN_X), F(A1_QN_G), F(A1_QN_B), 0, ws_xnq);
    k_ln_tok<<<TKV, 128, 0, stream>>>(ws_keyp, nullptr, F(A1_KN_G), F(A1_KN_B), 1, ws_xnk);
    k_ln_tok<<<TKV, 128, 0, stream>>>(ws_valp, nullptr, F(A1_VN_G), F(A1_VN_B), 1, ws_xnv);
    gemm(ws_xnq, w_a1q, F(A1_Q_B), 1.f, nullptr, 0, TQ1, DIMC, DIMC, 0, 0, nullptr, ws_Qp);
    gemm(ws_xnk, w_a1k, F(A1_K_B), 1.f, nullptr, 0, TKV, DIMC, DIMC, 0, 1, nullptr, ws_Kw);
    gemm(ws_xnv, w_a1v, F(A1_V_B), 1.f, nullptr, 0, TKV, DIMC, DIMC, 0, 3, nullptr, ws_Vt);
    k_attn<<<dim3(96, NHEADS, NWIN), 32, 0, stream>>>(ws_Qp, ws_Kw, ws_Vt, ws_att, 1);
    k_viewsum<<<TS, 128, 0, stream>>>(ws_att, NVIEW, ws_attsum);
    // z = mean_v(a @ p_w) + p_b + skip1(x, channel-major)
    gemm(ws_attsum, w_a1p, F(A1_P_B), 1.0f / NVIEW, F(IN_X), 1,
         TS, DIMC, DIMC, 0, 0, ws_qm, nullptr);
    // MLP 1
    k_ln_tok<<<TS, 128, 0, stream>>>(ws_qm, nullptr, F(P_PN1_G), F(P_PN1_B), 2, ws_xnt);
    gemm(ws_xnt, w_m1a, F(P_M1B1), 1.f, nullptr, 0, TS, DIMC, 256, 1, 0, nullptr, ws_mlph);
    gemm(ws_mlph, w_m1b, F(P_M1B2), 1.f, ws_qm, 0, TS, 256, DIMC, 0, 0, ws_qm, nullptr);

    // ---- pass 2 (queries broadcast over views -> single copy is exact) ----
    k_ln_tok<<<TS, 128, 0, stream>>>(ws_qm, nullptr, F(A2_QN_G), F(A2_QN_B), 2, ws_xnt);
    gemm(ws_xnt, w_a2q, F(A2_Q_B), 1.f, nullptr, 0, TS, DIMC, DIMC, 0, 0, nullptr, ws_Qp);
    k_ln_tok<<<TKV, 128, 0, stream>>>(ws_keyp, nullptr, F(A2_KN_G), F(A2_KN_B), 1, ws_xnk);
    k_ln_tok<<<TKV, 128, 0, stream>>>(ws_valp, nullptr, F(A2_VN_G), F(A2_VN_B), 1, ws_xnv);
    gemm(ws_xnk, w_a2k, F(A2_K_B), 1.f, nullptr, 0, TKV, DIMC, DIMC, 0, 2, nullptr, ws_Kw);
    gemm(ws_xnv, w_a2v, F(A2_V_B), 1.f, nullptr, 0, TKV, DIMC, DIMC, 0, 4, nullptr, ws_Vt);
    k_attn<<<dim3(16, NHEADS, NWIN), 32, 0, stream>>>(ws_Qp, ws_Kw, ws_Vt, ws_att, 2);
    k_viewsum<<<TS, 128, 0, stream>>>(ws_att, 1, ws_attsum);
    gemm(ws_attsum, w_a2p, F(A2_P_B), 1.0f, ws_qm, 0,
         TS, DIMC, DIMC, 0, 0, ws_qm2, nullptr);
    // MLP 2
    k_ln_tok<<<TS, 128, 0, stream>>>(ws_qm2, nullptr, F(P_PN2_G), F(P_PN2_B), 2, ws_xnt);
    gemm(ws_xnt, w_m2a, F(P_M2B1), 1.f, nullptr, 0, TS, DIMC, 256, 1, 0, nullptr, ws_mlph);
    gemm(ws_mlph, w_m2b, F(P_M2B2), 1.f, ws_qm2, 0, TS, 256, DIMC, 0, 0, ws_qm2, nullptr);

    // ---- final LayerNorm + transpose to (1,128,64,64) ----
    k_post_ln<<<TS, 128, 0, stream>>>(ws_qm2, F(P_POST_G), F(P_POST_B), (float*)d_out);
}
